// VirtualNodeMixin_33921651703943
// MI455X (gfx1250) — matrix operand order repaired and checked by reading, not compiled, not run on hardware
//
#include <hip/hip_runtime.h>

#define DIM 128
#define EPSF 1e-5f

typedef __attribute__((ext_vector_type(16))) _Float16 v16h;
typedef __attribute__((ext_vector_type(4)))  _Float16 v4h;
typedef __attribute__((ext_vector_type(8)))  float    v8f;
typedef __attribute__((ext_vector_type(4)))  float    v4f;

// ---------------- zero workspace ----------------
__global__ void zero_ws_kernel(float* __restrict__ ws, int n) {
    int i = (blockIdx.x * blockDim.x + threadIdx.x) * 4;
    if (i + 3 < n) {
        *(v4f*)(ws + i) = (v4f){0.f, 0.f, 0.f, 0.f};
    } else {
        for (int j = i; j < n; ++j) ws[j] = 0.f;
    }
}

// ---------------- segment-sum pooling (exploits sorted batch) ----------------
// 1 wave per block; lane t owns features 4t..4t+3 (one b128 per row per lane).
// 64 rows per block; run-length accumulate -> atomics only at batch-id changes.
__global__ __launch_bounds__(32)
void pool_kernel(const float* __restrict__ h, const int* __restrict__ batch,
                 float* __restrict__ pool, float* __restrict__ cnt, int N) {
    const int f4   = threadIdx.x * 4;
    const int base = blockIdx.x * 64;
    const int lim  = (N - base < 64) ? (N - base) : 64;

    v4f acc = {0.f, 0.f, 0.f, 0.f};
    float run = 0.f;
    int cur = -1;
    for (int r = 0; r < lim; ++r) {
        const int row = base + r;
        const int b = batch[row];            // wave-uniform load
        if (b != cur) {
            if (cur >= 0) {
                float* p = &pool[(size_t)cur * DIM + f4];
                __hip_atomic_fetch_add(p + 0, acc.x, __ATOMIC_RELAXED, __HIP_MEMORY_SCOPE_AGENT);
                __hip_atomic_fetch_add(p + 1, acc.y, __ATOMIC_RELAXED, __HIP_MEMORY_SCOPE_AGENT);
                __hip_atomic_fetch_add(p + 2, acc.z, __ATOMIC_RELAXED, __HIP_MEMORY_SCOPE_AGENT);
                __hip_atomic_fetch_add(p + 3, acc.w, __ATOMIC_RELAXED, __HIP_MEMORY_SCOPE_AGENT);
                if (threadIdx.x == 0)
                    __hip_atomic_fetch_add(&cnt[cur], run, __ATOMIC_RELAXED, __HIP_MEMORY_SCOPE_AGENT);
            }
            cur = b; acc = (v4f){0.f, 0.f, 0.f, 0.f}; run = 0.f;
        }
        const v4f hv = *(const v4f*)(h + (size_t)row * DIM + f4);
        acc += hv;
        run += 1.f;
    }
    if (cur >= 0) {
        float* p = &pool[(size_t)cur * DIM + f4];
        __hip_atomic_fetch_add(p + 0, acc.x, __ATOMIC_RELAXED, __HIP_MEMORY_SCOPE_AGENT);
        __hip_atomic_fetch_add(p + 1, acc.y, __ATOMIC_RELAXED, __HIP_MEMORY_SCOPE_AGENT);
        __hip_atomic_fetch_add(p + 2, acc.z, __ATOMIC_RELAXED, __HIP_MEMORY_SCOPE_AGENT);
        __hip_atomic_fetch_add(p + 3, acc.w, __ATOMIC_RELAXED, __HIP_MEMORY_SCOPE_AGENT);
        if (threadIdx.x == 0)
            __hip_atomic_fetch_add(&cnt[cur], run, __ATOMIC_RELAXED, __HIP_MEMORY_SCOPE_AGENT);
    }
}

// ---------------- fused VN MLP: mean + add + GEMM1 + LN + ReLU + GEMM2 ----------------
// 256 threads = 8 waves; block computes 16 output rows x 128 cols.
// Each wave: one 16x16 tile, K=128 as 4x v_wmma_f32_16x16x32_f16.
__global__ __launch_bounds__(256)
void vn_mlp_kernel(const float* __restrict__ pool, const float* __restrict__ cnt,
                   const float* __restrict__ vn_h,
                   const float* __restrict__ W1,  const float* __restrict__ b1,
                   const float* __restrict__ ln_g, const float* __restrict__ ln_b,
                   const float* __restrict__ W2,  const float* __restrict__ b2,
                   float* __restrict__ vn_out) {
    __shared__ _Float16 Wlds[DIM][DIM];   // 32 KB, staged W1 then reused for W2
    __shared__ _Float16 Afrag[16][DIM];   // 4 KB  f16 activations
    __shared__ float    Xb[16][DIM];      // 8 KB  GEMM1 output
    __shared__ float    mu_s[16], rs_s[16];

    const int tid  = threadIdx.x;
    const int wave = tid >> 5;
    const int lane = tid & 31;
    const int hh   = lane >> 4;           // half-lane (K split)
    const int l16  = lane & 15;           // row (A) / col (B,C,D) within tile
    const int r0   = blockIdx.x * 16;     // output row base
    const int colg = wave * 16 + l16;     // this lane's output column

    // Stage activations A = pooled/count + vn_h  (f16, b128 loads / b64 LDS stores)
    for (int e = tid * 4; e < 16 * DIM; e += 256 * 4) {
        const int r = e >> 7, c0 = e & 127;
        float cv = cnt[r0 + r]; cv = cv > 1.f ? cv : 1.f;
        const float inv = 1.f / cv;
        const v4f pv = *(const v4f*)(pool + (size_t)(r0 + r) * DIM + c0);
        const v4f vv = *(const v4f*)(vn_h + (size_t)(r0 + r) * DIM + c0);
        v4h o;
        o.x = (_Float16)(pv.x * inv + vv.x);
        o.y = (_Float16)(pv.y * inv + vv.y);
        o.z = (_Float16)(pv.z * inv + vv.z);
        o.w = (_Float16)(pv.w * inv + vv.w);
        *(v4h*)(&Afrag[r][c0]) = o;
    }
    // Stage W1 -> f16 LDS (vectorized)
    for (int e = tid * 4; e < DIM * DIM; e += 256 * 4) {
        const v4f w = *(const v4f*)(W1 + e);
        v4h o; o.x = (_Float16)w.x; o.y = (_Float16)w.y;
               o.z = (_Float16)w.z; o.w = (_Float16)w.w;
        *(v4h*)(&Wlds[e >> 7][e & 127]) = o;
    }
    __syncthreads();

    // ---- GEMM1: x = A @ W1^T + b1  (B-frag col n == W1 row n, contiguous) ----
    float bias = b1[colg];
    v8f c1 = {bias, bias, bias, bias, bias, bias, bias, bias};
    for (int kk = 0; kk < 4; ++kk) {
        const int k0 = kk * 32;
        v16h a, b;
        #pragma unroll
        for (int u = 0; u < 8; ++u) {
            // A k order used here (contiguous): lanes0-15 K=0-15 ; lanes16-31 K=16-31 (matches B)
            const int ak = hh * 16 + 2 * u;
            a[2*u]   = Afrag[l16][k0 + ak];
            a[2*u+1] = Afrag[l16][k0 + ak + 1];
            // B k order used here (contiguous): lanes0-15 K=0-15 ; lanes16-31 K=16-31 (2 per VGPR)
            const int bk = hh * 16 + 2 * u;
            b[2*u]   = Wlds[colg][k0 + bk];
            b[2*u+1] = Wlds[colg][k0 + bk + 1];
        }
        c1 = __builtin_amdgcn_wmma_f32_16x16x32_f16(false, a, false, b,
                                                    (short)0, c1, false, false);
    }
    // C/D layout: VGPR i -> row i + 8*half, col = l16 (+ wave*16)
    #pragma unroll
    for (int i = 0; i < 8; ++i) Xb[i + 8 * hh][colg] = c1[i];
    __syncthreads();   // Xb complete; all Wlds reads done

    // Stage W2 (reuse Wlds) while 16 threads compute LayerNorm stats
    for (int e = tid * 4; e < DIM * DIM; e += 256 * 4) {
        const v4f w = *(const v4f*)(W2 + e);
        v4h o; o.x = (_Float16)w.x; o.y = (_Float16)w.y;
               o.z = (_Float16)w.z; o.w = (_Float16)w.w;
        *(v4h*)(&Wlds[e >> 7][e & 127]) = o;
    }
    if (tid < 16) {
        float s = 0.f;
        for (int j = 0; j < DIM; j += 4) {
            const v4f x = *(const v4f*)(&Xb[tid][j]);
            s += x.x + x.y + x.z + x.w;
        }
        const float mu = s * (1.f / DIM);
        float v2 = 0.f;
        for (int j = 0; j < DIM; j += 4) {
            const v4f x = *(const v4f*)(&Xb[tid][j]);
            const v4f d = x - mu;
            v2 += d.x*d.x + d.y*d.y + d.z*d.z + d.w*d.w;
        }
        mu_s[tid] = mu;
        rs_s[tid] = rsqrtf(v2 * (1.f / DIM) + EPSF);
    }
    __syncthreads();

    // LN + ReLU -> f16 activations (reuse Afrag)
    for (int e = tid * 4; e < 16 * DIM; e += 256 * 4) {
        const int r = e >> 7, c0 = e & 127;
        const float mu = mu_s[r], rs = rs_s[r];
        const v4f x = *(const v4f*)(&Xb[r][c0]);
        const v4f g = *(const v4f*)(ln_g + c0);
        const v4f bb = *(const v4f*)(ln_b + c0);
        v4f y = (x - mu) * rs * g + bb;
        y.x = y.x > 0.f ? y.x : 0.f;
        y.y = y.y > 0.f ? y.y : 0.f;
        y.z = y.z > 0.f ? y.z : 0.f;
        y.w = y.w > 0.f ? y.w : 0.f;
        v4h o; o.x = (_Float16)y.x; o.y = (_Float16)y.y;
               o.z = (_Float16)y.z; o.w = (_Float16)y.w;
        *(v4h*)(&Afrag[r][c0]) = o;
    }
    __syncthreads();

    // ---- GEMM2: vn_out = relu(ln(x)) @ W2^T + b2 ----
    float bias2 = b2[colg];
    v8f c2 = {bias2, bias2, bias2, bias2, bias2, bias2, bias2, bias2};
    for (int kk = 0; kk < 4; ++kk) {
        const int k0 = kk * 32;
        v16h a, b;
        #pragma unroll
        for (int u = 0; u < 8; ++u) {
            const int ak = hh * 16 + 2 * u;
            a[2*u]   = Afrag[l16][k0 + ak];
            a[2*u+1] = Afrag[l16][k0 + ak + 1];
            const int bk = hh * 16 + 2 * u;
            b[2*u]   = Wlds[colg][k0 + bk];
            b[2*u+1] = Wlds[colg][k0 + bk + 1];
        }
        c2 = __builtin_amdgcn_wmma_f32_16x16x32_f16(false, a, false, b,
                                                    (short)0, c2, false, false);
    }
    #pragma unroll
    for (int i = 0; i < 8; ++i)
        vn_out[(size_t)(r0 + i + 8 * hh) * DIM + colg] = c2[i];
}

// ---------------- streaming scatter: h_out = h + vn_out[batch] ----------------
// One v4f per thread, fully coalesced b128 traffic. h is last-use -> NT load;
// h_out is write-only -> NT store (keeps L2 free for the hot 4MB vn_out table).
__global__ __launch_bounds__(256)
void scatter_kernel(const float* __restrict__ h, const int* __restrict__ batch,
                    const float* __restrict__ vn_out, float* __restrict__ h_out,
                    int N) {
    const long long t   = (long long)blockIdx.x * blockDim.x + threadIdx.x;
    const long long row = t >> 5;
    if (row >= N) return;
    const int f = (int)(t & 31) * 4;
    const int b = batch[row];
    const v4f hv = __builtin_nontemporal_load((const v4f*)(h + row * DIM + f));
    const v4f vv = *(const v4f*)(vn_out + (long long)b * DIM + f);
    const v4f o  = hv + vv;
    __builtin_nontemporal_store(o, (v4f*)(h_out + row * DIM + f));
}

extern "C" void kernel_launch(void* const* d_in, const int* in_sizes, int n_in,
                              void* d_out, int out_size, void* d_ws, size_t ws_size,
                              hipStream_t stream) {
    const float* h     = (const float*)d_in[0];
    const int*   batch = (const int*)d_in[1];
    const float* vn_h  = (const float*)d_in[2];
    const float* W1    = (const float*)d_in[3];
    const float* b1    = (const float*)d_in[4];
    const float* ln_g  = (const float*)d_in[5];
    const float* ln_b  = (const float*)d_in[6];
    const float* W2    = (const float*)d_in[7];
    const float* b2    = (const float*)d_in[8];

    const int N = in_sizes[0] / DIM;     // 500000
    const int B = in_sizes[2] / DIM;     // 8192

    float* pool  = (float*)d_ws;                 // [B, DIM]
    float* cnt   = pool + (size_t)B * DIM;       // [B]
    float* h_out = (float*)d_out;                // [N, DIM]
    float* vn_o  = h_out + (size_t)N * DIM;      // [B, DIM]

    const int zn = B * DIM + B;
    zero_ws_kernel<<<(zn / 4 + 255) / 256, 256, 0, stream>>>(pool, zn);

    pool_kernel<<<(N + 63) / 64, 32, 0, stream>>>(h, batch, pool, cnt, N);

    vn_mlp_kernel<<<B / 16, 256, 0, stream>>>(pool, cnt, vn_h, W1, b1,
                                              ln_g, ln_b, W2, b2, vn_o);

    const long long tot = (long long)N * 32;     // v4f threads
    scatter_kernel<<<(unsigned)((tot + 255) / 256), 256, 0, stream>>>(
        h, batch, vn_o, h_out, N);
}
